// Swin_41609643164065
// MI455X (gfx1250) — matrix operand order repaired and checked by reading, not compiled, not run on hardware
//
// Swin-V2-Tiny forward for MI455X (gfx1250, wave32, WMMA + TDM).
//
// Roofline: ~180 GFLOP of GEMM work vs ~0.5 GB of traffic @ 23.3 TB/s ->
// bandwidth/latency bound. Strategy:
//  * every matmul through v_wmma_f32_16x16x32_f16 (f16 in, f32 accumulate)
//  * GEMM operand staging via the Tensor Data Mover (TENSOR_LOAD_TO_LDS,
//    double-buffered, s_wait_tensorcnt) so each 64x32 / 16x32 tile is one
//    DMA descriptor instead of 48 scalar loads per wave
//  * fused window attention (QK^T -> scale/bias/mask -> softmax -> PV) in a
//    single wave32 kernel per (window, head); 64x64 scores live in LDS
//  * head_dim == 32 at every stage => one 16x16x32 WMMA spans the full K of
//    QK^T; window N == 64 => 4x4 score tiles, 4x2 PV tiles.

#include <hip/hip_runtime.h>
#include <hip/hip_bf16.h>
#include <math.h>

typedef __attribute__((ext_vector_type(16))) _Float16 v16h;
typedef __attribute__((ext_vector_type(8)))  float    v8f;
typedef unsigned int u32x4 __attribute__((ext_vector_type(4)));
typedef int          i32x4 __attribute__((ext_vector_type(4)));
typedef int          i32x8 __attribute__((ext_vector_type(8)));

#define WS 8
#define HD 32   // head_dim, constant across stages

// ---------------------------------------------------------------------------
// TDM: issue a 2-D TENSOR_LOAD_TO_LDS.  Descriptor per CDNA5 ISA:
//   group0: [1:0]=count=1, [63:32]=lds_addr, [120:64]=global_addr, [127:126]=2
//   group1: [17:16]=data_size(2 -> 4B), [79:48]=tensor_dim0, [111:80]=tensor_dim1,
//           [127:112]=tile_dim0, [143:128]=tile_dim1, [207:160]=tensor_dim0_stride
//   groups 2/3 zero (2-D tensor), cpol 0.
// All arguments are block-uniform -> SGPRs. Issued by one wave; TENSORcnt
// tracks completion on that wave only.
// ---------------------------------------------------------------------------
__device__ __forceinline__ void tdm_load_2d(const void* gaddr, unsigned lds_addr,
                                            unsigned rowlen_elems, unsigned nrows_tensor,
                                            unsigned tile0_elems, unsigned tile1_rows,
                                            unsigned stride0_elems)
{
  unsigned long long ga = (unsigned long long)gaddr;
  u32x4 g0;
  g0[0] = 1u;                                            // count=1, user mode
  g0[1] = lds_addr;                                      // LDS byte address
  g0[2] = (unsigned)(ga & 0xffffffffu);
  g0[3] = (unsigned)((ga >> 32) & 0x01ffffffu) | (2u << 30);  // addr[56:32] | type=2
  i32x8 g1;
  g1[0] = (int)(2u << 16);                               // data_size = 4 bytes
  g1[1] = (int)((rowlen_elems & 0xffffu) << 16);         // tensor_dim0[15:0]
  g1[2] = (int)((rowlen_elems >> 16) | ((nrows_tensor & 0xffffu) << 16));
  g1[3] = (int)((nrows_tensor >> 16) | (tile0_elems << 16));   // tile_dim0
  g1[4] = (int)tile1_rows;                               // tile_dim1 (tile_dim2 = 0)
  g1[5] = (int)stride0_elems;                            // tensor_dim0_stride lo
  g1[6] = 0;
  g1[7] = 0;
  i32x4 z4 = {0, 0, 0, 0};
#if defined(__clang_major__) && (__clang_major__ >= 23)
  i32x8 z8 = {0, 0, 0, 0, 0, 0, 0, 0};
  __builtin_amdgcn_tensor_load_to_lds(g0, g1, z4, z4, z8, 0);
#else
  __builtin_amdgcn_tensor_load_to_lds(g0, g1, z4, z4, 0);
#endif
}

// Flat address low 32 bits == LDS byte offset (aperture mapping, CDNA5 ISA)
__device__ __forceinline__ unsigned lds_off(const void* p)
{
  return (unsigned)(unsigned long long)p;
}

// ---------------------------------------------------------------------------
// TDM-staged GEMM: C[M,N] = act(A[M,K] @ W[N,K]^T + bias)
// Block = 4 waves = 64(M) x 16(N) tile. K stepped by 32; A tile 64x32 and
// B tile 16x32 staged to LDS by the TDM (double-buffered); transfer of chunk
// k+1 overlaps WMMA of chunk k. Requires M%64==0, N%16==0, K%32==0 (true for
// every GEMM in this net). act: 0 none, 1 exact GELU.
// ---------------------------------------------------------------------------
__global__ void __launch_bounds__(128)
gemm_wmma_tdm(const float* __restrict__ A, const float* __restrict__ W,
              const float* __restrict__ bias, float* __restrict__ C,
              int M, int N, int K, int act)
{
  __shared__ float tA[2][64 * 32];
  __shared__ float tB[2][16 * 32];

  const int lane = threadIdx.x;          // 0..31
  const int wave = threadIdx.y;          // 0..3
  const int tilesN = N >> 4;
  const int tm = (blockIdx.x / tilesN) << 6;
  const int tn = (blockIdx.x % tilesN) << 4;
  const int half = lane >> 4;
  const int l15  = lane & 15;
  const int row  = wave * 16 + l15;      // A row within the 64-row tile

  const float* Abase = A + (size_t)tm * K;
  const float* Wbase = W + (size_t)tn * K;
  const int nk = K >> 5;

  if (wave == 0) {
    tdm_load_2d(Abase, lds_off(tA[0]), (unsigned)K, 64u, 32u, 64u, (unsigned)K);
    tdm_load_2d(Wbase, lds_off(tB[0]), (unsigned)K, 16u, 32u, 16u, (unsigned)K);
  }

  v8f acc = {};
  for (int kt = 0; kt < nk; ++kt) {
    const int cur = kt & 1;
    if (wave == 0) __builtin_amdgcn_s_wait_tensorcnt(0);
    __syncthreads();
    if (wave == 0 && kt + 1 < nk) {
      const int k1 = (kt + 1) << 5;
      tdm_load_2d(Abase + k1, lds_off(tA[cur ^ 1]), (unsigned)K, 64u, 32u, 64u, (unsigned)K);
      tdm_load_2d(Wbase + k1, lds_off(tB[cur ^ 1]), (unsigned)K, 16u, 32u, 16u, (unsigned)K);
    }

    v16h a, b;
    {
      const float4* p = (const float4*)(tA[cur] + row * 32 + half * 16);
      const float4* q = (const float4*)(tA[cur] + row * 32 + half * 16 + 8);
      float4 x0 = p[0], x1 = p[1], x2 = q[0], x3 = q[1];
      a[0] = (_Float16)x0.x; a[1] = (_Float16)x0.y; a[2] = (_Float16)x0.z; a[3] = (_Float16)x0.w;
      a[4] = (_Float16)x1.x; a[5] = (_Float16)x1.y; a[6] = (_Float16)x1.z; a[7] = (_Float16)x1.w;
      a[8] = (_Float16)x2.x; a[9] = (_Float16)x2.y; a[10] = (_Float16)x2.z; a[11] = (_Float16)x2.w;
      a[12] = (_Float16)x3.x; a[13] = (_Float16)x3.y; a[14] = (_Float16)x3.z; a[15] = (_Float16)x3.w;
    }
    {
      const float4* p = (const float4*)(tB[cur] + l15 * 32 + half * 16);
      float4 y0 = p[0], y1 = p[1], y2 = p[2], y3 = p[3];
      b[0] = (_Float16)y0.x; b[1] = (_Float16)y0.y; b[2] = (_Float16)y0.z; b[3] = (_Float16)y0.w;
      b[4] = (_Float16)y1.x; b[5] = (_Float16)y1.y; b[6] = (_Float16)y1.z; b[7] = (_Float16)y1.w;
      b[8] = (_Float16)y2.x; b[9] = (_Float16)y2.y; b[10] = (_Float16)y2.z; b[11] = (_Float16)y2.w;
      b[12] = (_Float16)y3.x; b[13] = (_Float16)y3.y; b[14] = (_Float16)y3.z; b[15] = (_Float16)y3.w;
    }
    acc = __builtin_amdgcn_wmma_f32_16x16x32_f16(false, a, false, b,
                                                 (short)0, acc, false, false);
    __syncthreads();   // all reads of buf[cur] done before it is overwritten
  }

#pragma unroll
  for (int r = 0; r < 8; ++r) {
    const int m = tm + wave * 16 + r + 8 * half;   // C/D layout: lanes 16-31 = M+8
    const int n = tn + l15;
    float v = acc[r] + (bias ? bias[n] : 0.0f);
    if (act == 1) v = 0.5f * v * (1.0f + erff(v * 0.70710678118f));
    C[(size_t)m * N + n] = v;
  }
}

// ---------------------------------------------------------------------------
// Fused window attention, one wave32 block per (window, head).
// ---------------------------------------------------------------------------
__global__ void __launch_bounds__(32)
win_attn_wmma(const float* __restrict__ qkv, const float* __restrict__ bias,
              const float* __restrict__ lsc, float* __restrict__ out,
              int nH, int dim, int Hs, int shift)
{
  __shared__ _Float16 qs[64 * HD];
  __shared__ _Float16 ks[64 * HD];
  __shared__ _Float16 vs[64 * HD];
  __shared__ float    Sm[64 * 64];
  __shared__ _Float16 Pm[64 * 64];
  __shared__ int      rid[64];

  const int lane = threadIdx.x;
  const int h    = blockIdx.x % nH;
  const int win  = blockIdx.x / nH;
  const int nW   = Hs >> 3;
  const int winPerImg = nW * nW;
  const int C3   = 3 * dim;
  const size_t base = (size_t)win * 64 * C3;
  const float* q0 = qkv + base + h * HD;
  const float* k0 = qkv + base + dim + h * HD;
  const float* v0 = qkv + base + 2 * dim + h * HD;

  // Stage normalized q/k and raw v into LDS as f16 (float4 loads); region ids.
  for (int r = lane; r < 64; r += 32) {
    const float4* qr = (const float4*)(q0 + (size_t)r * C3);
    const float4* kr = (const float4*)(k0 + (size_t)r * C3);
    const float4* vr = (const float4*)(v0 + (size_t)r * C3);
    float4 qv[8], kv[8], vv[8];
    float nq = 0.f, nk = 0.f;
#pragma unroll
    for (int d = 0; d < 8; ++d) {
      qv[d] = qr[d]; kv[d] = kr[d]; vv[d] = vr[d];
      nq += qv[d].x * qv[d].x + qv[d].y * qv[d].y + qv[d].z * qv[d].z + qv[d].w * qv[d].w;
      nk += kv[d].x * kv[d].x + kv[d].y * kv[d].y + kv[d].z * kv[d].z + kv[d].w * kv[d].w;
    }
    nq = 1.0f / fmaxf(sqrtf(nq), 1e-12f);
    nk = 1.0f / fmaxf(sqrtf(nk), 1e-12f);
#pragma unroll
    for (int d = 0; d < 8; ++d) {
      qs[r * HD + 4 * d + 0] = (_Float16)(qv[d].x * nq);
      qs[r * HD + 4 * d + 1] = (_Float16)(qv[d].y * nq);
      qs[r * HD + 4 * d + 2] = (_Float16)(qv[d].z * nq);
      qs[r * HD + 4 * d + 3] = (_Float16)(qv[d].w * nq);
      ks[r * HD + 4 * d + 0] = (_Float16)(kv[d].x * nk);
      ks[r * HD + 4 * d + 1] = (_Float16)(kv[d].y * nk);
      ks[r * HD + 4 * d + 2] = (_Float16)(kv[d].z * nk);
      ks[r * HD + 4 * d + 3] = (_Float16)(kv[d].w * nk);
      vs[r * HD + 4 * d + 0] = (_Float16)vv[d].x;
      vs[r * HD + 4 * d + 1] = (_Float16)vv[d].y;
      vs[r * HD + 4 * d + 2] = (_Float16)vv[d].z;
      vs[r * HD + 4 * d + 3] = (_Float16)vv[d].w;
    }
    int idv = 0;
    if (shift) {
      int wi = win % winPerImg;
      int wh = wi / nW, ww = wi % nW;
      int rr = r >> 3, cc = r & 7;
      int hsrc = (wh * 8 + rr + shift) % Hs;
      int wsrc = (ww * 8 + cc + shift) % Hs;
      int rh = (hsrc < Hs - 8) ? 0 : ((hsrc < Hs - shift) ? 1 : 2);
      int rw = (wsrc < Hs - 8) ? 0 : ((wsrc < Hs - shift) ? 1 : 2);
      idv = rh * 3 + rw;
    }
    rid[r] = idv;
  }
  // Single wave: LDS ops are in program order; no barrier needed.

  const int half = lane >> 4;
  const int l15  = lane & 15;
  const float sc = expf(fminf(lsc[h], 4.60517018599f));  // ln(100)

  // S = (q_n k_n^T) * sc + bias + mask   (4x4 tiles, K=32 in one WMMA)
  for (int mt = 0; mt < 4; ++mt) {
    for (int nt = 0; nt < 4; ++nt) {
      v16h a, b;
      v8f  acc = {};
      const int m = mt * 16 + l15;
#pragma unroll
      for (int e = 0; e < 8; ++e) {
        a[e]     = qs[m * HD + half * 8 + e];
        a[8 + e] = qs[m * HD + 16 + half * 8 + e];
      }
      const int n = nt * 16 + l15;
#pragma unroll
      for (int e = 0; e < 16; ++e)
        b[e] = ks[n * HD + (e < 8 ? half * 8 + e : 16 + half * 8 + (e - 8))];   // B[k][n] = k_n[n][k]
      acc = __builtin_amdgcn_wmma_f32_16x16x32_f16(false, a, false, b,
                                                   (short)0, acc, false, false);
#pragma unroll
      for (int r = 0; r < 8; ++r) {
        const int mm = mt * 16 + r + 8 * half;
        const int nn = nt * 16 + l15;
        float bi = bias[((size_t)h * 64 + mm) * 64 + nn];
        float mk = (rid[mm] == rid[nn]) ? 0.0f : -100.0f;
        Sm[mm * 64 + nn] = acc[r] * sc + bi + mk;
      }
    }
  }

  // Row softmax (each lane owns 2 rows), result as f16 in Pm.
  for (int r = lane; r < 64; r += 32) {
    float mx = -1e30f;
    for (int j = 0; j < 64; ++j) mx = fmaxf(mx, Sm[r * 64 + j]);
    float sum = 0.f;
    for (int j = 0; j < 64; ++j) { float e = expf(Sm[r * 64 + j] - mx); Sm[r * 64 + j] = e; sum += e; }
    float inv = 1.0f / sum;
    for (int j = 0; j < 64; ++j) Pm[r * 64 + j] = (_Float16)(Sm[r * 64 + j] * inv);
  }

  // O = P @ V   (4x2 tiles, K=64 -> two WMMA steps)
  for (int mt = 0; mt < 4; ++mt) {
    for (int nt = 0; nt < 2; ++nt) {
      v8f acc = {};
#pragma unroll
      for (int kt = 0; kt < 2; ++kt) {
        v16h a, b;
        const int m  = mt * 16 + l15;
        const int kb = kt * 32;
#pragma unroll
        for (int e = 0; e < 8; ++e) {
          a[e]     = Pm[m * 64 + kb + half * 16 + e];
          a[8 + e] = Pm[m * 64 + kb + half * 16 + 8 + e];
        }
#pragma unroll
        for (int e = 0; e < 16; ++e)
          b[e] = vs[(kb + half * 16 + e) * HD + nt * 16 + l15];
        acc = __builtin_amdgcn_wmma_f32_16x16x32_f16(false, a, false, b,
                                                     (short)0, acc, false, false);
      }
#pragma unroll
      for (int r = 0; r < 8; ++r) {
        const int mm = mt * 16 + r + 8 * half;
        out[((size_t)win * 64 + mm) * dim + h * HD + nt * 16 + l15] = acc[r];
      }
    }
  }
}

// ---------------------------------------------------------------------------
// Patch embed: out[t, n] = sum_{c,i,j} x[b,c,4ph+i,4pw+j] * w[n, c*16+i*4+j] + b[n]
// ---------------------------------------------------------------------------
__global__ void patch_embed(const float* __restrict__ x, const float* __restrict__ w,
                            const float* __restrict__ bias, float* __restrict__ out, int B)
{
  int id = blockIdx.x * blockDim.x + threadIdx.x;
  int total = B * 64 * 64 * 96;
  if (id >= total) return;
  int n = id % 96;
  int t = id / 96;
  int pw = t % 64; int tmp = t / 64; int ph = tmp % 64; int b = tmp / 64;
  const float* xb = x + (size_t)b * 3 * 256 * 256;
  float acc = bias[n];
  for (int c = 0; c < 3; ++c)
    for (int i = 0; i < 4; ++i)
      for (int j = 0; j < 4; ++j)
        acc += xb[((size_t)c * 256 + ph * 4 + i) * 256 + pw * 4 + j] *
               w[n * 48 + c * 16 + i * 4 + j];
  out[(size_t)t * 96 + n] = acc;
}

// ---------------------------------------------------------------------------
// Row LayerNorm kernels
// ---------------------------------------------------------------------------
__global__ void ln_rows(const float* __restrict__ in, float* __restrict__ out,
                        const float* __restrict__ g, const float* __restrict__ b,
                        int T, int C)
{
  int t = blockIdx.x * blockDim.x + threadIdx.x;
  if (t >= T) return;
  const float* r = in + (size_t)t * C;
  float m = 0.f; for (int c = 0; c < C; ++c) m += r[c]; m /= C;
  float v = 0.f; for (int c = 0; c < C; ++c) { float d = r[c] - m; v += d * d; } v /= C;
  float rs = rsqrtf(v + 1e-5f);
  float* o = out + (size_t)t * C;
  for (int c = 0; c < C; ++c) o[c] = (r[c] - m) * rs * g[c] + b[c];
}

__global__ void add_ln_rows(float* __restrict__ h, const float* __restrict__ y,
                            const float* __restrict__ g, const float* __restrict__ b,
                            int T, int C)
{
  int t = blockIdx.x * blockDim.x + threadIdx.x;
  if (t >= T) return;
  const float* r = y + (size_t)t * C;
  float m = 0.f; for (int c = 0; c < C; ++c) m += r[c]; m /= C;
  float v = 0.f; for (int c = 0; c < C; ++c) { float d = r[c] - m; v += d * d; } v /= C;
  float rs = rsqrtf(v + 1e-5f);
  float* o = h + (size_t)t * C;
  for (int c = 0; c < C; ++c) o[c] += (r[c] - m) * rs * g[c] + b[c];
}

// h(b,hh,ww) += ln(y[window-ordered row mapped from (hh,ww) incl. un-roll])
__global__ void unpart_add_ln(float* __restrict__ h, const float* __restrict__ y,
                              const float* __restrict__ g, const float* __restrict__ b,
                              int B, int Hs, int C, int shift)
{
  int t = blockIdx.x * blockDim.x + threadIdx.x;
  int T = B * Hs * Hs;
  if (t >= T) return;
  int ww = t % Hs; int tmp = t / Hs; int hh = tmp % Hs; int bb = tmp / Hs;
  int a  = (hh - shift + Hs) % Hs;
  int c2 = (ww - shift + Hs) % Hs;
  int nW = Hs >> 3;
  size_t row = (((size_t)bb * nW + (a >> 3)) * nW + (c2 >> 3)) * 64 + (a & 7) * 8 + (c2 & 7);
  const float* r = y + row * C;
  float m = 0.f; for (int c = 0; c < C; ++c) m += r[c]; m /= C;
  float v = 0.f; for (int c = 0; c < C; ++c) { float d = r[c] - m; v += d * d; } v /= C;
  float rs = rsqrtf(v + 1e-5f);
  float* o = h + (size_t)t * C;
  for (int c = 0; c < C; ++c) o[c] += (r[c] - m) * rs * g[c] + b[c];
}

// Window partition with cyclic shift (roll by -shift).
__global__ void win_part(const float* __restrict__ h, float* __restrict__ xw,
                         int B, int Hs, int C, int shift)
{
  size_t id = (size_t)blockIdx.x * blockDim.x + threadIdx.x;
  size_t total = (size_t)B * Hs * Hs * C;
  if (id >= total) return;
  int c = (int)(id % C); size_t t = id / C;
  int tok = (int)(t & 63); size_t wn = t >> 6;
  int nW = Hs >> 3;
  int wwi = (int)(wn % nW); size_t tmp = wn / nW;
  int whi = (int)(tmp % nW); int bb = (int)(tmp / nW);
  int r = tok >> 3, cc = tok & 7;
  int hs = (whi * 8 + r + shift) % Hs;
  int ws2 = (wwi * 8 + cc + shift) % Hs;
  xw[id] = h[(((size_t)bb * Hs + hs) * Hs + ws2) * C + c];
}

// Patch-merge gather: xcat[b,h2,w2, q*C + c] = h[b, 2h2+(q&1), 2w2+(q>>1), c]
__global__ void merge_build(const float* __restrict__ h, float* __restrict__ xcat,
                            int B, int Hs /*old*/, int C /*old*/)
{
  size_t id = (size_t)blockIdx.x * blockDim.x + threadIdx.x;
  int H2 = Hs >> 1;
  size_t total = (size_t)B * H2 * H2 * 4 * C;
  if (id >= total) return;
  int c4 = (int)(id % (4 * C)); size_t t = id / (4 * C);
  int w2 = (int)(t % H2); size_t tmp = t / H2;
  int h2 = (int)(tmp % H2); int bb = (int)(tmp / H2);
  int q = c4 / C, c = c4 % C;
  int dh = q & 1, dw = q >> 1;
  xcat[id] = h[(((size_t)bb * Hs + 2 * h2 + dh) * Hs + 2 * w2 + dw) * C + c];
}

// ---------------------------------------------------------------------------
// Continuous position bias MLP (2 -> 512 -> nH), then 16*sigmoid, gathered.
// ---------------------------------------------------------------------------
__device__ inline float cpb_coord(int i)
{
  float x = (float)(i - 7) * (8.0f / 7.0f);
  float s = (x > 0.f) ? 1.0f : ((x < 0.f) ? -1.0f : 0.0f);
  return s * log2f(fabsf(x) + 1.0f) * (1.0f / 3.0f);
}

__global__ void cpb_hidden_k(const float* __restrict__ w1, const float* __restrict__ b1,
                             float* __restrict__ hid)
{
  int id = blockIdx.x * blockDim.x + threadIdx.x;
  if (id >= 225 * 512) return;
  int u = id % 512; int t = id / 512;
  float v = cpb_coord(t / 15) * w1[u * 2] + cpb_coord(t % 15) * w1[u * 2 + 1] + b1[u];
  hid[id] = fmaxf(v, 0.0f);
}

__global__ void cpb_bias_k(const float* __restrict__ hid, const float* __restrict__ w2,
                           float* __restrict__ bias, int nH)
{
  int id = blockIdx.x * blockDim.x + threadIdx.x;
  if (id >= nH * 4096) return;
  int nm = id % 4096; int h = id / 4096;
  int n = nm >> 6, m = nm & 63;
  int t = ((n >> 3) - (m >> 3) + 7) * 15 + ((n & 7) - (m & 7) + 7);
  const float* hr = hid + (size_t)t * 512;
  const float* wr = w2 + (size_t)h * 512;
  float s = 0.f;
  for (int u = 0; u < 512; ++u) s += hr[u] * wr[u];
  bias[id] = 16.0f / (1.0f + expf(-s));
}

// ---------------------------------------------------------------------------
// Host orchestration
// ---------------------------------------------------------------------------
extern "C" void kernel_launch(void* const* d_in, const int* in_sizes, int n_in,
                              void* d_out, int out_size, void* d_ws, size_t ws_size,
                              hipStream_t stream)
{
  (void)in_sizes; (void)n_in; (void)out_size; (void)ws_size;
  const int B = 16;
  const int DEPTHS[4] = {2, 2, 6, 2};
  const int HEADS[4]  = {3, 6, 12, 24};

  // ---- flatten inputs: x first, then params leaves (pytree order:
  //      dict keys alphabetical, lists in index order) ----
  const float* x = (const float*)d_in[0];
  int cur = 1;
  auto nxt = [&]() -> const float* { return (const float*)d_in[cur++]; };

  const float* patch_b   = nxt();  // 'b'
  const float* patch_lnb = nxt();  // 'ln_b'
  const float* patch_lng = nxt();  // 'ln_g'
  const float* patch_w   = nxt();  // 'w'

  struct Blk {
    const float *cpb_b1, *cpb_w1, *cpb_w2;
    const float *ln1_b, *ln1_g, *ln2_b, *ln2_g;
    const float *lsc;
    const float *mlp_b1, *mlp_b2, *mlp_w1, *mlp_w2;
    const float *proj_b, *proj_w, *qkv_b, *qkv_w;
  };
  struct Stg { Blk blk[6]; const float *m_lnb, *m_lng, *m_red; };
  Stg st[4];
  for (int i = 0; i < 4; ++i) {
    for (int j = 0; j < DEPTHS[i]; ++j) {
      Blk& Bk = st[i].blk[j];
      Bk.cpb_b1 = nxt(); Bk.cpb_w1 = nxt(); Bk.cpb_w2 = nxt();
      Bk.ln1_b = nxt(); Bk.ln1_g = nxt(); Bk.ln2_b = nxt(); Bk.ln2_g = nxt();
      Bk.lsc = nxt();
      Bk.mlp_b1 = nxt(); Bk.mlp_b2 = nxt(); Bk.mlp_w1 = nxt(); Bk.mlp_w2 = nxt();
      Bk.proj_b = nxt(); Bk.proj_w = nxt(); Bk.qkv_b = nxt(); Bk.qkv_w = nxt();
    }
    if (i > 0) { st[i].m_lnb = nxt(); st[i].m_lng = nxt(); st[i].m_red = nxt(); }
  }

  // ---- workspace carve-up (floats) ----
  float* wsf = (float*)d_ws;
  size_t off = 0;
  auto carve = [&](size_t n) { float* p = wsf + off; off += (n + 63) & ~(size_t)63; return p; };
  float* buf_h   = carve((size_t)6291456);   // activations [T, C]        (stage-1 max)
  float* buf_xw  = carve((size_t)6291456);   // window-partitioned / attn-out
  float* buf_tmp = carve((size_t)6291456);   // proj/mlp2/merge output
  float* buf_qkv = carve((size_t)18874368);  // [T, 3C]
  float* buf_mlp = carve((size_t)25165824);  // [T, 4C]
  float* buf_hid = carve((size_t)225 * 512); // cpb hidden
  float* buf_bia = carve((size_t)24 * 4096); // cpb bias [nH,64,64]

  auto gemm = [&](const float* A, const float* W, const float* bias, float* C,
                  int M, int N, int K, int act) {
    int blocks = (M >> 6) * (N >> 4);      // 64x16 tile per 4-wave block
    gemm_wmma_tdm<<<dim3(blocks), dim3(32, 4), 0, stream>>>(A, W, bias, C, M, N, K, act);
  };
  auto ceil256 = [](size_t n) { return (unsigned)((n + 255) / 256); };

  // ---- patch embed + LN ----
  {
    size_t total = (size_t)B * 64 * 64 * 96;
    patch_embed<<<ceil256(total), 256, 0, stream>>>(x, patch_w, patch_b, buf_tmp, B);
    int T = B * 64 * 64;
    ln_rows<<<ceil256(T), 256, 0, stream>>>(buf_tmp, buf_h, patch_lng, patch_lnb, T, 96);
  }

  size_t outOff = 0;  // bytes into d_out
  for (int i = 0; i < 4; ++i) {
    const int Hs  = 64 >> i;
    const int dim = 96 << i;
    const int nH  = HEADS[i];
    const int T   = B * Hs * Hs;
    const int nWinTot = T / 64;

    if (i > 0) {
      // patch merging: gather 2x2 -> GEMM red_w (K = 2*dim) -> LN
      const int HsP = Hs * 2, dimP = dim / 2;
      size_t total = (size_t)T * 4 * dimP;
      merge_build<<<ceil256(total), 256, 0, stream>>>(buf_h, buf_xw, B, HsP, dimP);
      gemm(buf_xw, st[i].m_red, nullptr, buf_tmp, T, dim, 2 * dim, 0);
      ln_rows<<<ceil256(T), 256, 0, stream>>>(buf_tmp, buf_h, st[i].m_lng, st[i].m_lnb, T, dim);
    }

    for (int j = 0; j < DEPTHS[i]; ++j) {
      const Blk& Bk = st[i].blk[j];
      const int shift = ((j & 1) && Hs > WS) ? (WS / 2) : 0;

      // CPB relative-position bias
      cpb_hidden_k<<<ceil256(225 * 512), 256, 0, stream>>>(Bk.cpb_w1, Bk.cpb_b1, buf_hid);
      cpb_bias_k<<<ceil256((size_t)nH * 4096), 256, 0, stream>>>(buf_hid, Bk.cpb_w2, buf_bia, nH);

      // window partition (+roll), qkv GEMM, fused attention
      size_t total = (size_t)T * dim;
      win_part<<<ceil256(total), 256, 0, stream>>>(buf_h, buf_xw, B, Hs, dim, shift);
      gemm(buf_xw, Bk.qkv_w, Bk.qkv_b, buf_qkv, T, 3 * dim, dim, 0);
      win_attn_wmma<<<dim3(nWinTot * nH), dim3(32), 0, stream>>>(
          buf_qkv, buf_bia, Bk.lsc, buf_xw, nH, dim, Hs, shift);

      // proj, then residual + post-LN with window un-partition (+un-roll)
      gemm(buf_xw, Bk.proj_w, Bk.proj_b, buf_tmp, T, dim, dim, 0);
      unpart_add_ln<<<ceil256(T), 256, 0, stream>>>(buf_h, buf_tmp, Bk.ln1_g, Bk.ln1_b,
                                                    B, Hs, dim, shift);

      // MLP (GELU) + residual + post-LN
      gemm(buf_h, Bk.mlp_w1, Bk.mlp_b1, buf_mlp, T, 4 * dim, dim, 1);
      gemm(buf_mlp, Bk.mlp_w2, Bk.mlp_b2, buf_tmp, T, dim, 4 * dim, 0);
      add_ln_rows<<<ceil256(T), 256, 0, stream>>>(buf_h, buf_tmp, Bk.ln2_g, Bk.ln2_b, T, dim);
    }

    // append stage output
    size_t bytes = (size_t)T * dim * sizeof(float);
    hipMemcpyAsync((char*)d_out + outOff, buf_h, bytes, hipMemcpyDeviceToDevice, stream);
    outOff += bytes;
  }
}
